// VanilaRNN_65017214926936
// MI455X (gfx1250) — matrix operand order repaired and checked by reading, not compiled, not run on hardware
//
#include <hip/hip_runtime.h>
#include <math.h>

// ---------------------------------------------------------------------------
// VanillaRNN for MI455X (gfx1250): bf16 WMMA GEMMs + persistent recurrence
//   B=64, T=512, INPUT=512, HIDDEN=1024, OUTPUT=512
// ---------------------------------------------------------------------------

#define BATCH   64
#define TSTEPS  512
#define INPUT_D 512
#define HIDDEN_D 1024
#define OUTPUT_D 512

typedef __bf16 bf16_t;
typedef __bf16 v16bf __attribute__((ext_vector_type(16)));
typedef float  v8f   __attribute__((ext_vector_type(8)));

union FragBF { uint4 q[2]; v16bf v; };

// round-to-nearest-even f32 -> bf16 (inputs are finite; skip NaN path)
__device__ __forceinline__ unsigned short f2bf(float f) {
  unsigned u = __float_as_uint(f);
  unsigned r = u + 0x7FFFu + ((u >> 16) & 1u);
  return (unsigned short)(r >> 16);
}

// A fragment (16x32 bf16, M x K), A row-major with leading dim lda.
// Lane m in [0,16): rows; lane>>4 selects K-half. Two 16B chunks per lane.
__device__ __forceinline__ v16bf load_afrag(const bf16_t* A, int lda, int m0,
                                            int k0, int lane) {
  const int m = lane & 15, kh = lane >> 4;
  const bf16_t* p = A + (size_t)(m0 + m) * lda + (k0 + kh * 8);
  FragBF f;
  f.q[0] = *(const uint4*)(p);        // K = base+0..7
  f.q[1] = *(const uint4*)(p + 16);   // K = base+16..23
  return f.v;
}

// B fragment (32x16 bf16, K x N) from BT = B transposed (N x K row-major).
// Lane n in [0,16): columns; lane>>4 selects K-half. Two 16B chunks per lane.
__device__ __forceinline__ v16bf load_bfrag(const bf16_t* BT, int ldb, int n0,
                                            int k0, int lane) {
  const int n = lane & 15, kh = lane >> 4;
  const bf16_t* p = BT + (size_t)(n0 + n) * ldb + (k0 + kh * 8);
  FragBF f;
  f.q[0] = *(const uint4*)(p);
  f.q[1] = *(const uint4*)(p + 16);
  return f.v;
}

__device__ __forceinline__ v8f wmma_bf16(v16bf a, v16bf b, v8f c) {
  // D = A x B + C, f32 accumulate, 16x16x32 bf16
  return __builtin_amdgcn_wmma_f32_16x16x32_bf16(
      /*neg_a=*/false, a, /*neg_b=*/false, b,
      /*c_mod=*/(short)0, c, /*reuse_a=*/false, /*reuse_b=*/false);
}

// ---------------------------------------------------------------------------
// Conversion kernels (run once per launch; all weights fit in L2)
// ---------------------------------------------------------------------------
__global__ void cvt_bf16(const float* __restrict__ src,
                         unsigned short* __restrict__ dst, int n) {
  int i = blockIdx.x * blockDim.x + threadIdx.x;
  if (i < n) dst[i] = f2bf(src[i]);
}

// dst[c*R + r] = bf16(src[r*C + c])  (store weight transposed, K-major)
__global__ void cvt_bf16_t(const float* __restrict__ src,
                           unsigned short* __restrict__ dst, int R, int C) {
  int i = blockIdx.x * blockDim.x + threadIdx.x;
  if (i < R * C) {
    int r = i / C, c = i % C;
    dst[(size_t)c * R + r] = f2bf(src[i]);
  }
}

// ---------------------------------------------------------------------------
// Phase 1: xin[b*T+t, :] = x @ W_in + b_in   (M=32768, K=512, N=1024)
// 64x64 tile per wave, 8 waves per block -> 1024 blocks.
// ---------------------------------------------------------------------------
__global__ __launch_bounds__(256) void gemm_xin(
    const bf16_t* __restrict__ xb, const bf16_t* __restrict__ WinT,
    const float* __restrict__ bin, float* __restrict__ xin) {
  const int lane = threadIdx.x & 31;
  const int wave = (threadIdx.x >> 5) + (blockIdx.x << 3);
  const int tile_m = wave >> 4;      // 16 tiles along N
  const int tile_n = wave & 15;
  const int m0 = tile_m * 64;
  const int n0 = tile_n * 64;

  v8f acc[4][4] = {};
#pragma unroll 1
  for (int k0 = 0; k0 < INPUT_D; k0 += 32) {
    // prefetch next K-slab of the streamed x operand
    if (k0 + 32 < INPUT_D)
      __builtin_prefetch(xb + (size_t)(m0 + (lane & 15)) * INPUT_D + k0 + 32, 0, 0);
    v16bf a[4], b[4];
#pragma unroll
    for (int i = 0; i < 4; ++i)
      a[i] = load_afrag(xb, INPUT_D, m0 + 16 * i, k0, lane);
#pragma unroll
    for (int j = 0; j < 4; ++j)
      b[j] = load_bfrag(WinT, INPUT_D, n0 + 16 * j, k0, lane);
#pragma unroll
    for (int i = 0; i < 4; ++i)
#pragma unroll
      for (int j = 0; j < 4; ++j)
        acc[i][j] = wmma_bf16(a[i], b[j], acc[i][j]);
  }

  const int n = lane & 15;
  const int mb = (lane >> 4) * 8;
#pragma unroll
  for (int j = 0; j < 4; ++j) {
    const int ncol = n0 + j * 16 + n;
    const float bias = bin[ncol];
#pragma unroll
    for (int i = 0; i < 4; ++i)
#pragma unroll
      for (int r = 0; r < 8; ++r)
        xin[(size_t)(m0 + i * 16 + mb + r) * HIDDEN_D + ncol] =
            acc[i][j][r] + bias;
  }
}

// ---------------------------------------------------------------------------
// Phase 2: persistent recurrence. 16 WGs x 4 waves = 64 waves; wave w owns
// columns [16w,16w+16) of H and computes the full 64x16 tile per step.
// Device-wide barrier: one counter per timestep (zeroed each launch).
// ---------------------------------------------------------------------------
#define RNN_WGS 16

__global__ __launch_bounds__(128) void rnn_recurrence(
    const float* __restrict__ xin, const bf16_t* __restrict__ WhT,
    const float* __restrict__ bh, bf16_t* h0, bf16_t* h1,
    unsigned* __restrict__ counters) {
  const int lane = threadIdx.x & 31;
  const int wave = (threadIdx.x >> 5) + (blockIdx.x << 2);
  const int n0 = wave * 16;
  const int n = lane & 15;
  const int mb = (lane >> 4) * 8;
  const int ncol = n0 + n;
  const float bias = bh[ncol];

  for (int t = 0; t < TSTEPS; ++t) {
    const bf16_t* hs = (t & 1) ? h1 : h0;
    bf16_t* hd = (t & 1) ? h0 : h1;

    v8f acc[4] = {};
#pragma unroll 1
    for (int k0 = 0; k0 < HIDDEN_D; k0 += 32) {
      v16bf b = load_bfrag(WhT, HIDDEN_D, n0, k0, lane);
      v16bf a0 = load_afrag(hs, HIDDEN_D, 0, k0, lane);
      v16bf a1 = load_afrag(hs, HIDDEN_D, 16, k0, lane);
      v16bf a2 = load_afrag(hs, HIDDEN_D, 32, k0, lane);
      v16bf a3 = load_afrag(hs, HIDDEN_D, 48, k0, lane);
      acc[0] = wmma_bf16(a0, b, acc[0]);
      acc[1] = wmma_bf16(a1, b, acc[1]);
      acc[2] = wmma_bf16(a2, b, acc[2]);
      acc[3] = wmma_bf16(a3, b, acc[3]);
    }

    // epilogue: h_new = tanh(xin_t + h@W_h + b_h), stored bf16
#pragma unroll
    for (int mt = 0; mt < 4; ++mt)
#pragma unroll
      for (int r = 0; r < 8; ++r) {
        const int m = mt * 16 + mb + r;  // batch index
        float v = acc[mt][r] + bias +
                  xin[((size_t)m * TSTEPS + t) * HIDDEN_D + ncol];
        ((unsigned short*)hd)[(size_t)m * HIDDEN_D + ncol] = f2bf(tanhf(v));
      }

    // device-wide barrier: release stores, arrive, spin, acquire
    __syncthreads();
    __threadfence();
    if (threadIdx.x == 0) {
      atomicAdd(&counters[t], 1u);
      while (atomicAdd(&counters[t], 0u) < (unsigned)RNN_WGS)
        __builtin_amdgcn_s_sleep(1);
    }
    __syncthreads();
    __threadfence();
  }
}

// ---------------------------------------------------------------------------
// Phase 3: out = h_final @ W_out + b_out  (M=64, K=1024, N=512)
// 32 waves (8 WGs x 4 waves), 64x16 tile each.  h_final lives in h0
// (T=512 even -> last step t=511 writes h0).
// ---------------------------------------------------------------------------
__global__ __launch_bounds__(128) void gemm_out(
    const bf16_t* __restrict__ h, const bf16_t* __restrict__ WoutT,
    const float* __restrict__ bout, float* __restrict__ out) {
  const int lane = threadIdx.x & 31;
  const int wave = (threadIdx.x >> 5) + (blockIdx.x << 2);
  const int n0 = wave * 16;
  const int n = lane & 15;
  const int mb = (lane >> 4) * 8;
  const int ncol = n0 + n;
  const float bias = bout[ncol];

  v8f acc[4] = {};
#pragma unroll 1
  for (int k0 = 0; k0 < HIDDEN_D; k0 += 32) {
    v16bf b = load_bfrag(WoutT, HIDDEN_D, n0, k0, lane);
    v16bf a0 = load_afrag(h, HIDDEN_D, 0, k0, lane);
    v16bf a1 = load_afrag(h, HIDDEN_D, 16, k0, lane);
    v16bf a2 = load_afrag(h, HIDDEN_D, 32, k0, lane);
    v16bf a3 = load_afrag(h, HIDDEN_D, 48, k0, lane);
    acc[0] = wmma_bf16(a0, b, acc[0]);
    acc[1] = wmma_bf16(a1, b, acc[1]);
    acc[2] = wmma_bf16(a2, b, acc[2]);
    acc[3] = wmma_bf16(a3, b, acc[3]);
  }
#pragma unroll
  for (int mt = 0; mt < 4; ++mt)
#pragma unroll
    for (int r = 0; r < 8; ++r) {
      const int m = mt * 16 + mb + r;
      out[(size_t)m * OUTPUT_D + ncol] = acc[mt][r] + bias;
    }
}

// ---------------------------------------------------------------------------
// Host launcher
// ---------------------------------------------------------------------------
extern "C" void kernel_launch(void* const* d_in, const int* in_sizes, int n_in,
                              void* d_out, int out_size, void* d_ws,
                              size_t ws_size, hipStream_t stream) {
  const float* x     = (const float*)d_in[0];  // [64,512,512]
  const float* W_in  = (const float*)d_in[1];  // [512,1024]
  const float* b_in  = (const float*)d_in[2];  // [1024]
  const float* W_h   = (const float*)d_in[3];  // [1024,1024]
  const float* b_h   = (const float*)d_in[4];  // [1024]
  const float* W_out = (const float*)d_in[5];  // [1024,512]
  const float* b_out = (const float*)d_in[6];  // [512]
  float* out = (float*)d_out;                  // [64,512]

  char* ws = (char*)d_ws;
  size_t off = 0;
  auto carve = [&](size_t bytes) {
    void* p = ws + off;
    off += (bytes + 255) & ~(size_t)255;
    return p;
  };

  const size_t n_x   = (size_t)BATCH * TSTEPS * INPUT_D;   // 16.7M
  const size_t n_xin = (size_t)BATCH * TSTEPS * HIDDEN_D;  // 33.5M

  unsigned short* xb    = (unsigned short*)carve(n_x * 2);                   // 32 MB
  unsigned short* WinT  = (unsigned short*)carve((size_t)INPUT_D * HIDDEN_D * 2);
  unsigned short* WhT   = (unsigned short*)carve((size_t)HIDDEN_D * HIDDEN_D * 2);
  unsigned short* WoutT = (unsigned short*)carve((size_t)HIDDEN_D * OUTPUT_D * 2);
  float*          xin   = (float*)carve(n_xin * 4);                          // 128 MB
  unsigned short* h0    = (unsigned short*)carve((size_t)BATCH * HIDDEN_D * 2);
  unsigned short* h1    = (unsigned short*)carve((size_t)BATCH * HIDDEN_D * 2);
  unsigned*       ctrs  = (unsigned*)carve(TSTEPS * sizeof(unsigned));

  // zero-init recurrence state + barrier counters (re-done on every replay)
  hipMemsetAsync(h0, 0, (size_t)BATCH * HIDDEN_D * 2, stream);
  hipMemsetAsync(ctrs, 0, TSTEPS * sizeof(unsigned), stream);

  // precision conversions
  cvt_bf16<<<(unsigned)((n_x + 255) / 256), 256, 0, stream>>>(x, xb, (int)n_x);
  cvt_bf16_t<<<(INPUT_D * HIDDEN_D + 255) / 256, 256, 0, stream>>>(
      W_in, WinT, INPUT_D, HIDDEN_D);
  cvt_bf16_t<<<(HIDDEN_D * HIDDEN_D + 255) / 256, 256, 0, stream>>>(
      W_h, WhT, HIDDEN_D, HIDDEN_D);
  cvt_bf16_t<<<(HIDDEN_D * OUTPUT_D + 255) / 256, 256, 0, stream>>>(
      W_out, WoutT, HIDDEN_D, OUTPUT_D);

  // phase 1: input projection for all timesteps (bandwidth-bound GEMM)
  gemm_xin<<<1024, 256, 0, stream>>>((const bf16_t*)xb, (const bf16_t*)WinT,
                                     b_in, xin);

  // phase 2: persistent latency-bound recurrence (single kernel, 512 steps)
  rnn_recurrence<<<RNN_WGS, 128, 0, stream>>>(
      xin, (const bf16_t*)WhT, b_h, (bf16_t*)h0, (bf16_t*)h1, ctrs);

  // phase 3: output projection (h_final is in h0 after an even step count)
  gemm_out<<<8, 128, 0, stream>>>((const bf16_t*)h0, (const bf16_t*)WoutT,
                                  b_out, out);
}
